// ZSinkhornHead_14319420965445
// MI455X (gfx1250) — compile-verified, matrix operand order checked by reading, not run on hardware
//
#include <hip/hip_runtime.h>
#include <hip/hip_bf16.h>
#include <stdint.h>

#define VOCAB 50257
#define DIM   768
#define ROWS  4096           // B*S = 2*2048
#define INV_EPS 20.0f        // 1 / 0.05
#define SINK_ITERS 5

typedef __bf16  v16bf __attribute__((ext_vector_type(16)));
typedef float   v8f   __attribute__((ext_vector_type(8)));

// ---------------------------------------------------------------------------
// Wave (32-lane) reductions
// ---------------------------------------------------------------------------
__device__ __forceinline__ float wave_max(float v) {
#pragma unroll
    for (int off = 16; off > 0; off >>= 1)
        v = fmaxf(v, __shfl_xor(v, off, 32));
    return v;
}
__device__ __forceinline__ float wave_sum(float v) {
#pragma unroll
    for (int off = 16; off > 0; off >>= 1)
        v += __shfl_xor(v, off, 32);
    return v;
}

// ---------------------------------------------------------------------------
// Kernel 1: per-row L2 normalize [rows, 768] fp32 -> bf16.  One wave per row.
// ---------------------------------------------------------------------------
__global__ __launch_bounds__(256) void norm_rows_kernel(const float* __restrict__ in,
                                                        __bf16* __restrict__ out,
                                                        int rows) {
    const int wavesPerBlock = blockDim.x >> 5;
    const int row  = blockIdx.x * wavesPerBlock + (threadIdx.x >> 5);
    const int lane = threadIdx.x & 31;
    if (row >= rows) return;

    const float* r = in + (size_t)row * DIM;
    float ss = 0.0f;
#pragma unroll 4
    for (int i = lane; i < DIM; i += 32) {
        float v = r[i];
        ss += v * v;
    }
    ss = wave_sum(ss);
    const float inv = 1.0f / fmaxf(sqrtf(ss), 1e-12f);

    __bf16* o = out + (size_t)row * DIM;
#pragma unroll 4
    for (int i = lane; i < DIM; i += 32)
        o[i] = (__bf16)(r[i] * inv);
}

// ---------------------------------------------------------------------------
// WMMA fragment load for 16-bit A/B (16x32 tile, MxK), ISA 7.12.2 layout:
//   lanes 0-15  (M = lane)    : elems 0..7 = K 0..7,  elems 8..15 = K 16..23
//   lanes 16-31 (M = lane-16) : elems 0..7 = K 8..15, elems 8..15 = K 24..31
// ---------------------------------------------------------------------------
union BFrag {
    v16bf v;
    uint4 q[2];
};

__device__ __forceinline__ v16bf load_frag_row(const __bf16* __restrict__ rowp,
                                               int k0, int lane) {
    const int lo = (lane & 16) ? 8 : 0;
    const __bf16* p = rowp + k0 + lo;                       // 16B aligned
    BFrag f;
    f.q[0] = *(const uint4*)(p);
    f.q[1] = *(const uint4*)(p + 16);
    return f.v;
}

// ---------------------------------------------------------------------------
// Kernel 2: logits[4096, 50257] = (Xn @ Wn^T) * 20,  bf16 WMMA, fp32 accum.
// Block = 256 threads = 8 waves -> 128x128 tile (wave grid 2x4, 64x32/wave).
// ---------------------------------------------------------------------------
__global__ __launch_bounds__(256) void gemm_kernel(const __bf16* __restrict__ Xn,
                                                   const __bf16* __restrict__ Wn,
                                                   float* __restrict__ out) {
    const int lane = threadIdx.x & 31;
    const int wave = threadIdx.x >> 5;      // 0..7
    const int wm   = wave >> 2;             // 0..1
    const int wn   = wave & 3;              // 0..3
    const int m0   = blockIdx.y * 128 + wm * 64;
    const int n0   = blockIdx.x * 128 + wn * 32;
    const int lm   = lane & 15;

    // Hoist per-lane row pointers (B row clamp is loop-invariant).
    const __bf16* arow[4];
#pragma unroll
    for (int mi = 0; mi < 4; ++mi)
        arow[mi] = Xn + (size_t)(m0 + mi * 16 + lm) * DIM;

    const __bf16* brow[2];
#pragma unroll
    for (int ni = 0; ni < 2; ++ni) {
        int r = n0 + ni * 16 + lm;
        if (r >= VOCAB) r = VOCAB - 1;      // clamp: garbage cols never stored
        brow[ni] = Wn + (size_t)r * DIM;
    }

    v8f acc[4][2] = {};

    for (int k0 = 0; k0 < DIM; k0 += 32) {
        v16bf a[4], b[2];
#pragma unroll
        for (int ni = 0; ni < 2; ++ni)
            b[ni] = load_frag_row(brow[ni], k0, lane);
#pragma unroll
        for (int mi = 0; mi < 4; ++mi)
            a[mi] = load_frag_row(arow[mi], k0, lane);
#pragma unroll
        for (int mi = 0; mi < 4; ++mi)
#pragma unroll
            for (int ni = 0; ni < 2; ++ni)
                acc[mi][ni] = __builtin_amdgcn_wmma_f32_16x16x32_bf16(
                    false, a[mi], false, b[ni], (short)0, acc[mi][ni], false, false);
    }

    // Epilogue: C/D layout -> VGPR r: lanes 0-15 are M=r, lanes 16-31 are M=r+8
    const int rowAdd = (lane & 16) ? 8 : 0;
#pragma unroll
    for (int mi = 0; mi < 4; ++mi) {
#pragma unroll
        for (int ni = 0; ni < 2; ++ni) {
            const int col = n0 + ni * 16 + lm;
            if (col < VOCAB) {
                const int rbase = m0 + mi * 16 + rowAdd;
#pragma unroll
                for (int r = 0; r < 8; ++r)
                    out[(size_t)(rbase + r) * VOCAB + col] = acc[mi][ni][r] * INV_EPS;
            }
        }
    }
}

// ---------------------------------------------------------------------------
// Kernel 3: 5x log_softmax over V, entirely in LDS (201 KB row fits in the
// 320 KB CDNA5 workgroup LDS), then + bias.  One 1024-thread block per row.
// Row staged with CDNA5 async global->LDS copies (ASYNCcnt, bypasses VGPRs).
// ---------------------------------------------------------------------------
#define NT 1024

__device__ __forceinline__ float block_max(float v, float* red) {
    __syncthreads();                       // protect red reuse (WAR)
    v = wave_max(v);
    const int w = threadIdx.x >> 5, l = threadIdx.x & 31;
    if (l == 0) red[w] = v;
    __syncthreads();
    if (w == 0) {
        float r = (l < (NT >> 5)) ? red[l] : -INFINITY;
        r = wave_max(r);
        if (l == 0) red[0] = r;
    }
    __syncthreads();
    return red[0];
}

__device__ __forceinline__ float block_sum(float v, float* red) {
    __syncthreads();
    v = wave_sum(v);
    const int w = threadIdx.x >> 5, l = threadIdx.x & 31;
    if (l == 0) red[w] = v;
    __syncthreads();
    if (w == 0) {
        float r = (l < (NT >> 5)) ? red[l] : 0.0f;
        r = wave_sum(r);
        if (l == 0) red[0] = r;
    }
    __syncthreads();
    return red[0];
}

__global__ __launch_bounds__(NT) void sinkhorn_kernel(float* __restrict__ logits,
                                                      const float* __restrict__ bias) {
    extern __shared__ float rowbuf[];          // VOCAB floats (~201 KB)
    __shared__ float red[32];

    float* g = logits + (size_t)blockIdx.x * VOCAB;
    const int tid = threadIdx.x;

    // --- Async global -> LDS stage (GLOBAL_LOAD_ASYNC_TO_LDS_B32, ASYNCcnt) ---
    for (int i = tid; i < VOCAB; i += NT) {
        uint32_t ldsAddr = (uint32_t)(uintptr_t)&rowbuf[i];
        const float* gp  = g + i;
        asm volatile("global_load_async_to_lds_b32 %0, %1, off"
                     :: "v"(ldsAddr), "v"(gp)
                     : "memory");
    }
    asm volatile("s_wait_asynccnt 0x0" ::: "memory");
    __syncthreads();

    for (int it = 0; it < SINK_ITERS; ++it) {
        float m = -INFINITY;
        for (int i = tid; i < VOCAB; i += NT) m = fmaxf(m, rowbuf[i]);
        m = block_max(m, red);

        float s = 0.0f;
        for (int i = tid; i < VOCAB; i += NT) s += __expf(rowbuf[i] - m);
        s = block_sum(s, red);

        const float lz = m + __logf(s);
        __syncthreads();
        for (int i = tid; i < VOCAB; i += NT) rowbuf[i] -= lz;
        __syncthreads();
    }

    for (int i = tid; i < VOCAB; i += NT) g[i] = rowbuf[i] + bias[i];
}

// ---------------------------------------------------------------------------
extern "C" void kernel_launch(void* const* d_in, const int* in_sizes, int n_in,
                              void* d_out, int out_size, void* d_ws, size_t ws_size,
                              hipStream_t stream) {
    const float* x    = (const float*)d_in[0];   // [4096, 768]
    const float* w    = (const float*)d_in[1];   // [50257, 768]
    const float* bias = (const float*)d_in[2];   // [50257]
    float* out = (float*)d_out;                  // [4096, 50257]

    __bf16* Xn = (__bf16*)d_ws;                          // 6.3 MB
    __bf16* Wn = Xn + (size_t)ROWS * DIM;                // 77.2 MB (fits 192MB L2)

    // 1) L2-normalize into bf16 (8 waves / block -> 8 rows / block)
    norm_rows_kernel<<<ROWS / 8, 256, 0, stream>>>(x, Xn, ROWS);
    norm_rows_kernel<<<(VOCAB + 7) / 8, 256, 0, stream>>>(w, Wn, VOCAB);

    // 2) bf16 WMMA GEMM, 128x128 tiles
    dim3 grid((VOCAB + 127) / 128, ROWS / 128);
    gemm_kernel<<<grid, 256, 0, stream>>>(Xn, Wn, out);

    // 3) 5x log_softmax in LDS + bias, one block per row
    const size_t ldsBytes = (size_t)VOCAB * sizeof(float);
    sinkhorn_kernel<<<ROWS, NT, ldsBytes, stream>>>(out, bias);
}
